// BiDirectionalAttentionFlow_19335942766716
// MI455X (gfx1250) — matrix operand order repaired and checked by reading, not compiled, not run on hardware
//
#include <hip/hip_runtime.h>
#include <math.h>

// ---------------- WMMA types (CDNA5 / gfx1250, wave32) ----------------
typedef __attribute__((ext_vector_type(16))) _Float16 v16h;
typedef __attribute__((ext_vector_type(8)))  _Float16 v8h;
typedef __attribute__((ext_vector_type(4)))  _Float16 v4h;
typedef __attribute__((ext_vector_type(8)))  float    v8f;

union F16V { v16h v; v8h h[2]; };

// A 16x32 f16 fragment (ISA 7.12.2): two contiguous 8-half groups per lane:
//   j=0..7  -> k = (hi?8:0)+j ; j=8..15 -> k = 16+(hi?8:0)+(j-8)
__device__ __forceinline__ v16h mk_afrag(const _Float16* __restrict__ row, bool hi)
{
    F16V u;
    const int o = hi ? 8 : 0;
    u.h[0] = *(const v8h*)(row + o);
    u.h[1] = *(const v8h*)(row + o + 16);
    return u.v;
}
// B 32x16 f16 fragment from a k-contiguous B^T row: two contiguous 8-half groups:
//   j=0..7  -> k = (hi?8:0)+j ; j=8..15 -> k = 16+(hi?8:0)+(j-8)
__device__ __forceinline__ v16h mk_bfrag(const _Float16* __restrict__ row, bool hi)
{
    F16V u;
    const int o = hi ? 8 : 0;
    u.h[0] = *(const v8h*)(row + o);
    u.h[1] = *(const v8h*)(row + o + 16);
    return u.v;
}

__device__ __forceinline__ float sigf(float x) { return 1.0f / (1.0f + expf(-x)); }

// =====================================================================
// LDS-staged WMMA GEMM:  C[b] = A[b] @ op(B[b]) (+bias +bias2), A scaled
// by ascale[k].  A: [M,K] f32 row-major.  B: b_is_nk ? [N,K] : [K,N].
// Block tile 64x64, 4 waves, each wave a 32x32 subtile (2x2 wmma frags).
// Interior tiles stage via coalesced float4 global loads -> packed v4h
// LDS stores; edges fall back to guarded scalar staging.  Fragment
// reads are contiguous 16B LDS vector loads (ds_load_b128).
// =====================================================================
#define BM 64
#define BN 64
#define BK 32

__global__ void __launch_bounds__(128)
k_gemm(const float* __restrict__ A, const float* __restrict__ Bm,
       float* __restrict__ Cm, int M, int N, int K,
       long sA, long sB, long sC,
       const float* __restrict__ bias, const float* __restrict__ bias2,
       const float* __restrict__ ascale, int b_is_nk)
{
    __shared__ _Float16 As[BM][BK];   // [m][k]
    __shared__ _Float16 Bs[BN][BK];   // [n][k]  (B^T tile, k contiguous)

    int batch = blockIdx.z;
    A  += (long)batch * sA;
    Bm += (long)batch * sB;
    Cm += (long)batch * sC;

    int nbn = (N + BN - 1) / BN;
    int m0 = (blockIdx.x / nbn) * BM;
    int n0 = (blockIdx.x % nbn) * BN;

    int tid  = threadIdx.x;
    int wave = tid >> 5;
    int lane = tid & 31;
    int lr   = lane & 15;
    bool hi  = lane >= 16;
    int wm = (wave >> 1) * 32;        // wave 2x2 grid inside 64x64 block
    int wn = (wave & 1) * 32;

    const bool k4      = (K & 3) == 0;
    const bool fullMA  = (m0 + BM <= M);
    const bool fullNB  = (n0 + BN <= N);

    v8f acc00 = {}, acc01 = {}, acc10 = {}, acc11 = {};

    for (int k0 = 0; k0 < K; k0 += BK) {
        const bool fullK = (k0 + BK <= K);
        // ---- stage A tile ----
        if (fullMA && fullK && k4) {            // fast: float4 loads, v4h stores
            for (int q = tid; q < BM * 8; q += 128) {
                int mi = q >> 3, kk = (q & 7) * 4;
                float4 f = *(const float4*)(A + (long)(m0 + mi) * K + k0 + kk);
                if (ascale) {
                    float4 s = *(const float4*)(ascale + k0 + kk);
                    f.x *= s.x; f.y *= s.y; f.z *= s.z; f.w *= s.w;
                }
                v4h hv;
                hv[0] = (_Float16)f.x; hv[1] = (_Float16)f.y;
                hv[2] = (_Float16)f.z; hv[3] = (_Float16)f.w;
                *(v4h*)&As[mi][kk] = hv;
            }
        } else {                                 // edge: guarded scalar
            for (int idx = tid; idx < BM * BK; idx += 128) {
                int mi = idx >> 5, kk = idx & 31;
                int gm = m0 + mi, gk = k0 + kk;
                float v = 0.0f;
                if (gm < M && gk < K) {
                    v = A[(long)gm * K + gk];
                    if (ascale) v *= ascale[gk];
                }
                As[mi][kk] = (_Float16)v;
            }
        }
        // ---- stage B^T tile ----
        if (b_is_nk && fullNB && fullK && k4) {  // fast: float4 along k
            for (int q = tid; q < BN * 8; q += 128) {
                int ni = q >> 3, kk = (q & 7) * 4;
                float4 f = *(const float4*)(Bm + (long)(n0 + ni) * K + k0 + kk);
                v4h hv;
                hv[0] = (_Float16)f.x; hv[1] = (_Float16)f.y;
                hv[2] = (_Float16)f.z; hv[3] = (_Float16)f.w;
                *(v4h*)&Bs[ni][kk] = hv;
            }
        } else {
            for (int idx = tid; idx < BN * BK; idx += 128) {
                int ni, kk;
                if (b_is_nk) { ni = idx >> 5; kk = idx & 31; }  // coalesce on k
                else         { kk = idx >> 6; ni = idx & 63; }  // coalesce on n
                int gn = n0 + ni, gk = k0 + kk;
                float v = 0.0f;
                if (gn < N && gk < K)
                    v = b_is_nk ? Bm[(long)gn * K + gk] : Bm[(long)gk * N + gn];
                Bs[ni][kk] = (_Float16)v;
            }
        }
        __syncthreads();

        v16h a0 = mk_afrag(&As[wm + lr][0],      hi);
        v16h a1 = mk_afrag(&As[wm + 16 + lr][0], hi);
        v16h b0 = mk_bfrag(&Bs[wn + lr][0],      hi);
        v16h b1 = mk_bfrag(&Bs[wn + 16 + lr][0], hi);
        acc00 = __builtin_amdgcn_wmma_f32_16x16x32_f16(false, a0, false, b0, (short)0, acc00, false, false);
        acc01 = __builtin_amdgcn_wmma_f32_16x16x32_f16(false, a0, false, b1, (short)0, acc01, false, false);
        acc10 = __builtin_amdgcn_wmma_f32_16x16x32_f16(false, a1, false, b0, (short)0, acc10, false, false);
        acc11 = __builtin_amdgcn_wmma_f32_16x16x32_f16(false, a1, false, b1, (short)0, acc11, false, false);
        __syncthreads();
    }

    // ---- store (C/D layout: VGPR j -> M = j (+8 for hi half), N = lane&15) ----
    v8f* accs[4] = { &acc00, &acc01, &acc10, &acc11 };
#pragma unroll
    for (int mi2 = 0; mi2 < 2; ++mi2)
#pragma unroll
        for (int ni2 = 0; ni2 < 2; ++ni2) {
            v8f& a = *accs[mi2 * 2 + ni2];
#pragma unroll
            for (int j = 0; j < 8; ++j) {
                int m = m0 + wm + mi2 * 16 + j + (hi ? 8 : 0);
                int n = n0 + wn + ni2 * 16 + lr;
                if (m < M && n < N) {
                    float v = a[j];
                    if (bias)  v += bias[n];
                    if (bias2) v += bias2[n];
                    Cm[(long)m * N + n] = v;
                }
            }
        }
}

// =====================================================================
// Persistent-workgroup LSTM scan (one direction).
// Whh(f16), h(f16), c, g live in LDS (~175 KB of the 320 KB WGP budget).
// Per timestep: g = h @ Whh^T via WMMA (M=32,N=400,K=128 zero-padded),
// then fused i,f,g,o gate math; h written to the bilstm output slice.
// =====================================================================
__global__ void __launch_bounds__(512)
k_lstm_scan(const float* __restrict__ xg,   // [B,T,400]
            const float* __restrict__ Whh,  // [400,100] f32
            float* __restrict__ out,        // [B,T,200]
            int T, int outOffset, int reverse)
{
    extern __shared__ char smem[];
    float*     g   = (float*)smem;                  // 32*400 f32   (51200 B)
    float*     c   = g + 32 * 400;                  // 32*100 f32   (12800 B)
    _Float16*  whh = (_Float16*)(c + 32 * 100);     // 400*128 f16  (102400 B)
    _Float16*  h   = whh + 400 * 128;               // 32*128 f16   (8192 B)

    int tid = threadIdx.x, nthr = blockDim.x;
    for (int idx = tid; idx < 400 * 128; idx += nthr) {
        int n = idx >> 7, k = idx & 127;
        whh[idx] = (_Float16)((k < 100) ? Whh[n * 100 + k] : 0.0f);
    }
    for (int idx = tid; idx < 32 * 128; idx += nthr) h[idx] = (_Float16)0.0f;
    for (int idx = tid; idx < 32 * 100; idx += nthr) c[idx] = 0.0f;
    __syncthreads();

    int wave = tid >> 5, nw = nthr >> 5;
    int lane = tid & 31, lr = lane & 15;
    bool hi = lane >= 16;

    for (int s = 0; s < T; ++s) {
        int t = reverse ? (T - 1 - s) : s;
        // ---- g = h @ Whh^T  (2 x 25 tiles of 16x16, K = 128 in 4 chunks) ----
        for (int tile = wave; tile < 50; tile += nw) {       // wave-uniform
            int mt = tile & 1, nt = tile >> 1;
            const _Float16* hrow = h   + (mt * 16 + lr) * 128;
            const _Float16* wrow = whh + (nt * 16 + lr) * 128;
            v8f acc = {};
#pragma unroll
            for (int k0 = 0; k0 < 128; k0 += 32) {
                v16h a = mk_afrag(hrow + k0, hi);
                v16h b = mk_bfrag(wrow + k0, hi);
                acc = __builtin_amdgcn_wmma_f32_16x16x32_f16(false, a, false, b,
                                                             (short)0, acc, false, false);
            }
#pragma unroll
            for (int j = 0; j < 8; ++j)
                g[(mt * 16 + j + (hi ? 8 : 0)) * 400 + nt * 16 + lr] = acc[j];
        }
        __syncthreads();
        // ---- fused gates (PyTorch order i,f,g,o) ----
        for (int idx = tid; idx < 3200; idx += nthr) {
            int bb = idx / 100, n = idx % 100;
            const float* xr = xg + ((long)bb * T + t) * 400;
            float gi = g[bb * 400 + n]       + xr[n];
            float gf = g[bb * 400 + 100 + n] + xr[100 + n];
            float gg = g[bb * 400 + 200 + n] + xr[200 + n];
            float go = g[bb * 400 + 300 + n] + xr[300 + n];
            float cn = sigf(gf) * c[idx] + sigf(gi) * tanhf(gg);
            float hn = sigf(go) * tanhf(cn);
            c[idx] = cn;
            h[bb * 128 + n] = (_Float16)hn;
            out[((long)bb * T + t) * 200 + outOffset + n] = hn;
        }
        __syncthreads();
    }
}

// ---------------- small elementwise / gather kernels ----------------
__global__ void k_word_embed(const int* __restrict__ tok, const float* __restrict__ emb,
                             float* __restrict__ out, int nrows)
{
    int i = blockIdx.x * blockDim.x + threadIdx.x;
    if (i >= nrows * 100) return;
    int n = i / 100, d = i % 100;
    out[(long)n * 200 + d] = emb[(long)tok[n] * 100 + d];
}

__global__ void k_char_embed(const int* __restrict__ tok, const float* __restrict__ cemb,
                             const float* __restrict__ cw, const float* __restrict__ cb,
                             float* __restrict__ out)
{
    __shared__ float x[8][16];
    int n = blockIdx.x, tid = threadIdx.x;
    if (tid < 16) {
        int tk = tok[(long)n * 16 + tid];
        for (int cc = 0; cc < 8; ++cc) x[cc][tid] = cemb[(long)tk * 8 + cc];
    }
    __syncthreads();
    if (tid < 100) {
        float m = -1e30f;
        for (int t = 0; t < 12; ++t) {
            float acc = cb[tid];
            for (int cc = 0; cc < 8; ++cc)
#pragma unroll
                for (int k = 0; k < 5; ++k)
                    acc += x[cc][t + k] * cw[tid * 40 + cc * 5 + k];
            m = fmaxf(m, acc);
        }
        out[(long)n * 200 + 100 + tid] = fmaxf(m, 0.0f);   // max(relu) == relu(max)
    }
}

__global__ void k_highway_elem(float* __restrict__ x, const float* __restrict__ flow,
                               const float* __restrict__ gate, int n)
{
    int i = blockIdx.x * blockDim.x + threadIdx.x;
    if (i >= n) return;
    float gv = sigf(gate[i]);
    x[i] = gv * fmaxf(flow[i], 0.0f) + (1.0f - gv) * x[i];
}

__global__ void k_rowdot(const float* __restrict__ X, const float* __restrict__ w,
                         float* __restrict__ out, int rows, int K)
{
    int r = blockIdx.x * blockDim.x + threadIdx.x;
    if (r >= rows) return;
    float s = 0.0f;
    for (int k = 0; k < K; ++k) s += X[(long)r * K + k] * w[k];
    out[r] = s;
}

__global__ void k_softmax_q(const float* __restrict__ S, const float* __restrict__ cw1,
                            const float* __restrict__ qw2, float* __restrict__ attq,
                            float* __restrict__ smax)
{
    int r = blockIdx.x * blockDim.x + threadIdx.x;   // r = b*400 + i
    if (r >= 32 * 400) return;
    int b = r / 400;
    float v[50], c1 = cw1[r], m = -1e30f;
    for (int j = 0; j < 50; ++j) { v[j] = S[(long)r * 50 + j] + c1 + qw2[b * 50 + j]; m = fmaxf(m, v[j]); }
    smax[r] = m;
    float sum = 0.0f;
    for (int j = 0; j < 50; ++j) { v[j] = expf(v[j] - m); sum += v[j]; }
    float inv = 1.0f / sum;
    for (int j = 0; j < 50; ++j) attq[(long)r * 50 + j] = v[j] * inv;
}

__global__ void k_softmax_c(const float* __restrict__ smax, float* __restrict__ attc)
{
    __shared__ float red[256];
    int b = blockIdx.x, tid = threadIdx.x;
    float m = -1e30f;
    for (int i = tid; i < 400; i += 256) m = fmaxf(m, smax[b * 400 + i]);
    red[tid] = m; __syncthreads();
    for (int s = 128; s > 0; s >>= 1) { if (tid < s) red[tid] = fmaxf(red[tid], red[tid + s]); __syncthreads(); }
    m = red[0]; __syncthreads();
    float sum = 0.0f;
    for (int i = tid; i < 400; i += 256) sum += expf(smax[b * 400 + i] - m);
    red[tid] = sum; __syncthreads();
    for (int s = 128; s > 0; s >>= 1) { if (tid < s) red[tid] += red[tid + s]; __syncthreads(); }
    float inv = 1.0f / red[0];
    for (int i = tid; i < 400; i += 256) attc[b * 400 + i] = expf(smax[b * 400 + i] - m) * inv;
}

__global__ void k_q2c(const float* __restrict__ attc, const float* __restrict__ C,
                      float* __restrict__ q2c)
{
    int r = blockIdx.x * blockDim.x + threadIdx.x;   // r = b*200 + d
    if (r >= 32 * 200) return;
    int b = r / 200, d = r % 200;
    float s = 0.0f;
    for (int i = 0; i < 400; ++i) s += attc[b * 400 + i] * C[((long)b * 400 + i) * 200 + d];
    q2c[r] = s;
}

__global__ void k_buildG(const float* __restrict__ C, const float* __restrict__ c2q,
                         const float* __restrict__ q2c, float* __restrict__ G)
{
    int i = blockIdx.x * blockDim.x + threadIdx.x;   // over 12800*200
    if (i >= 32 * 400 * 200) return;
    int row = i / 200, d = i % 200, b = row / 400;
    float cv = C[i], c2 = c2q[i], q2 = q2c[b * 200 + d];
    long base = (long)row * 800;
    G[base + d]       = cv;
    G[base + 200 + d] = c2;
    G[base + 400 + d] = cv * c2;
    G[base + 600 + d] = cv * q2;
}

__global__ void k_pred(const float* __restrict__ G, const float* __restrict__ Mx,
                       const float* __restrict__ w, float* __restrict__ out)
{
    int r = blockIdx.x * blockDim.x + threadIdx.x;   // r = b*400 + i
    if (r >= 32 * 400) return;
    float s = 0.0f;
    for (int d = 0; d < 800; ++d) s += G[(long)r * 800 + d] * w[d];
    for (int d = 0; d < 200; ++d) s += Mx[(long)r * 200 + d] * w[800 + d];
    out[r] = s;
}

// =====================================================================
// Host orchestration
// =====================================================================
extern "C" void kernel_launch(void* const* d_in, const int* in_sizes, int n_in,
                              void* d_out, int out_size, void* d_ws, size_t ws_size,
                              hipStream_t stream)
{
    (void)in_sizes; (void)n_in; (void)out_size; (void)ws_size;
    auto P = [&](int i) { return (const float*)d_in[i]; };
    // params flatten order: 0 word_emb, 1 char_emb, 2 conv_w, 3 conv_b,
    // 4 hw_flow_w, 5 hw_flow_b, 6 hw_gate_w, 7 hw_gate_b,
    // 8..15 ctx_lstm (fwd Wih,Whh,bih,bhh; bwd ...), 16 sim_w,
    // 17..24 model_lstm0, 25..32 model_lstm1, 33..40 end_lstm,
    // 41 w_begin, 42 w_end, 43 context, 44 question, 45 char_ctx, 46 char_q.
    const int* ctx_tok  = (const int*)d_in[43];
    const int* q_tok    = (const int*)d_in[44];
    const int* ctx_char = (const int*)d_in[45];
    const int* q_char   = (const int*)d_in[46];

    // ---- workspace bump allocator (f32 elements) ----
    float* base = (float*)d_ws;  size_t off = 0;
    auto alloc = [&](size_t n) { float* r = base + off; off += n; return r; };
    float* CIN  = alloc(2560000);   // [32,400,200]
    float* QIN  = alloc(320000);    // [32,50,200]
    float* Cb   = alloc(2560000);   // C
    float* Qb   = alloc(320000);    // Q
    float* XGF  = alloc(5120000);   // xg fwd (max [32,400,400])
    float* XGB  = alloc(5120000);   // xg bwd
    float* TMP1 = alloc(2560000);   // highway flow lin
    float* TMP2 = alloc(2560000);   // highway gate lin
    float* S    = alloc(640000);    // [32,400,50]
    float* ATTQ = alloc(640000);
    float* CW1  = alloc(12800);
    float* QW2  = alloc(1600);
    float* SMAX = alloc(12800);
    float* ATTC = alloc(12800);
    float* Q2C  = alloc(6400);
    float* C2Q  = alloc(2560000);
    float* G    = alloc(10240000);  // [32,400,800]
    float* M0   = alloc(2560000);
    float* Mb   = alloc(2560000);
    float* O2   = alloc(2560000);

    auto gemm = [&](const float* A, const float* Bm, float* Cm, int M, int N, int K,
                    long sA, long sB, long sC, int batches,
                    const float* bias, const float* bias2, const float* ascale, int b_is_nk) {
        int nb = ((M + BM - 1) / BM) * ((N + BN - 1) / BN);
        dim3 grid(nb, 1, batches);
        k_gemm<<<grid, 128, 0, stream>>>(A, Bm, Cm, M, N, K, sA, sB, sC,
                                         bias, bias2, ascale, b_is_nk);
    };

    size_t lstm_shm = 32 * 400 * 4 + 32 * 100 * 4 + 400 * 128 * 2 + 32 * 128 * 2; // 174592 B

    auto bilstm = [&](const float* X, int T, int inDim, int pbase, float* out) {
        gemm(X, P(pbase + 0), XGF, 32 * T, 400, inDim, 0, 0, 0, 1,
             P(pbase + 2), P(pbase + 3), nullptr, 1);
        gemm(X, P(pbase + 4), XGB, 32 * T, 400, inDim, 0, 0, 0, 1,
             P(pbase + 6), P(pbase + 7), nullptr, 1);
        k_lstm_scan<<<1, 512, lstm_shm, stream>>>(XGF, P(pbase + 1), out, T, 0,   0);
        k_lstm_scan<<<1, 512, lstm_shm, stream>>>(XGB, P(pbase + 5), out, T, 100, 1);
    };

    auto highway = [&](float* X, int rows) {
        for (int l = 0; l < 2; ++l) {
            gemm(X, P(4) + (long)l * 200 * 200, TMP1, rows, 200, 200, 0, 0, 0, 1,
                 P(5) + l * 200, nullptr, nullptr, 1);
            gemm(X, P(6) + (long)l * 200 * 200, TMP2, rows, 200, 200, 0, 0, 0, 1,
                 P(7) + l * 200, nullptr, nullptr, 1);
            int n = rows * 200;
            k_highway_elem<<<(n + 255) / 256, 256, 0, stream>>>(X, TMP1, TMP2, n);
        }
    };

    // ---- embeddings ----
    k_word_embed<<<(12800 * 100 + 255) / 256, 256, 0, stream>>>(ctx_tok, P(0), CIN, 12800);
    k_word_embed<<<(1600 * 100 + 255) / 256, 256, 0, stream>>>(q_tok, P(0), QIN, 1600);
    k_char_embed<<<12800, 128, 0, stream>>>(ctx_char, P(1), P(2), P(3), CIN);
    k_char_embed<<<1600, 128, 0, stream>>>(q_char, P(1), P(2), P(3), QIN);

    // ---- highway + contextual BiLSTM ----
    highway(CIN, 12800);
    highway(QIN, 1600);
    bilstm(CIN, 400, 200, 8, Cb);
    bilstm(QIN, 50, 200, 8, Qb);

    // ---- trilinear attention ----
    const float* simw = P(16);
    k_rowdot<<<(12800 + 255) / 256, 256, 0, stream>>>(Cb, simw, CW1, 12800, 200);
    k_rowdot<<<(1600 + 255) / 256, 256, 0, stream>>>(Qb, simw + 200, QW2, 1600, 200);
    gemm(Cb, Qb, S, 400, 50, 200, 400 * 200, 50 * 200, 400 * 50, 32,
         nullptr, nullptr, simw + 400, 1);                 // (C*w3) @ Q^T, batched
    k_softmax_q<<<(12800 + 255) / 256, 256, 0, stream>>>(S, CW1, QW2, ATTQ, SMAX);
    k_softmax_c<<<32, 256, 0, stream>>>(SMAX, ATTC);
    gemm(ATTQ, Qb, C2Q, 400, 200, 50, 400 * 50, 50 * 200, 400 * 200, 32,
         nullptr, nullptr, nullptr, 0);                    // attq @ Q, batched
    k_q2c<<<(6400 + 255) / 256, 256, 0, stream>>>(ATTC, Cb, Q2C);
    k_buildG<<<(2560000 + 255) / 256, 256, 0, stream>>>(Cb, C2Q, Q2C, G);

    // ---- modeling + end LSTMs ----
    bilstm(G, 400, 800, 17, M0);
    bilstm(M0, 400, 200, 25, Mb);
    bilstm(Mb, 400, 200, 33, O2);

    // ---- span predictions ----
    float* out = (float*)d_out;
    k_pred<<<(12800 + 255) / 256, 256, 0, stream>>>(G, Mb, P(41), out);
    k_pred<<<(12800 + 255) / 256, 256, 0, stream>>>(G, O2, P(42), out + 12800);
}
